// CausalSelfAttention_36730560315493
// MI455X (gfx1250) — matrix operand order repaired and checked by reading, not compiled, not run on hardware
//
#include <hip/hip_runtime.h>
#include <hip/hip_bf16.h>

// ---------------------------------------------------------------------------
// CDNA5 (gfx1250) causal self-attention, bf16 WMMA + async global->LDS copies.
// Shapes fixed by the reference: B=4, T=2048, C=2048, H=16, D=128.
// ---------------------------------------------------------------------------

#define BB 4
#define TT 2048
#define CC 2048
#define HH 16
#define DD 128

typedef float v8f   __attribute__((ext_vector_type(8)));
typedef __bf16 v16bf __attribute__((ext_vector_type(16)));

union FragBF {          // one 16x32 bf16 A-fragment or 32x16 bf16 B-fragment
    v16bf v;
    uint4 q[2];
};

__device__ __forceinline__ v8f vzero8() {
    v8f z;
#pragma unroll
    for (int i = 0; i < 8; ++i) z[i] = 0.0f;
    return z;
}

__device__ __forceinline__ v8f wmma_bf16(const v16bf& a, const v16bf& b, const v8f& c) {
    return __builtin_amdgcn_wmma_f32_16x16x32_bf16(
        false, a, false, b, (short)0, c, false, false);
}

// Async copy of 16 bytes global -> LDS (CDNA5, ASYNCcnt-tracked).
// LDS operand is the 32-bit LDS byte address (= low 32 bits of the generic
// pointer per ISA 10.2).
__device__ __forceinline__ void async_copy_b128(const void* gptr, void* lptr) {
    unsigned int lds = (unsigned int)(unsigned long long)(size_t)lptr;
    unsigned long long ga = (unsigned long long)(size_t)gptr;
    asm volatile("global_load_async_to_lds_b128 %0, %1, off"
                 :: "v"(lds), "v"(ga) : "memory");
}

__device__ __forceinline__ void wait_asynccnt_0() {
    asm volatile("s_wait_asynccnt 0x0" ::: "memory");
}
__device__ __forceinline__ void wait_asynccnt_4() {
    asm volatile("s_wait_asynccnt 0x4" ::: "memory");
}

// ---------------------------------------------------------------------------
// Kernel 1: fp32 -> bf16 conversion
// ---------------------------------------------------------------------------
__global__ void __launch_bounds__(256) cvt_f32_bf16(const float* __restrict__ in,
                                                    __bf16* __restrict__ out,
                                                    size_t n) {
    size_t i = (size_t)blockIdx.x * blockDim.x + threadIdx.x;
    if (i < n) out[i] = (__bf16)in[i];
}

// ---------------------------------------------------------------------------
// Kernel 2/5: bf16 GEMM  C[m][n] = sum_k A[m][k] * Bm[n][k].
// Block tile 128x128, K-step 32, 256 threads = 8 waves (2x4 wave grid),
// each wave computes 64x32 via 4x2 WMMA accumulators.
// K-tiles are double-buffered in LDS and filled with async global->LDS DMA.
// MODE 0: scatter bf16 q/k/v into [B,H,T,D] buffers (QKV projection).
// MODE 1: write fp32 C row-major (output projection).
// ---------------------------------------------------------------------------
template <int MODE>
__global__ void __launch_bounds__(256) gemm_bf16_wmma(
    const __bf16* __restrict__ A, const __bf16* __restrict__ Bm,
    float* __restrict__ Cout,
    __bf16* __restrict__ Qb, __bf16* __restrict__ Kb, __bf16* __restrict__ Vb,
    int M, int N, int K) {
    __shared__ __bf16 As[2][128][32];   // 16 KB
    __shared__ __bf16 Bs[2][128][32];   // 16 KB

    const int tid  = threadIdx.x;
    const int wave = tid >> 5;
    const int lane = tid & 31;
    const int l15  = lane & 15;
    const int lh   = lane >> 4;      // lane half: 0 or 1
    const int wm   = wave >> 2;      // 0..1  (rows of waves, 64 rows each)
    const int wn   = wave & 3;       // 0..3  (cols of waves, 32 cols each)

    const int n0 = blockIdx.x * 128;
    const int m0 = blockIdx.y * 128;

    // this thread's two 8-element chunks of each 128x32 tile
    const int idx0 = tid * 2;
    const int r0 = idx0 >> 2,      c0 = (idx0 & 3) * 8;
    const int r1 = (idx0 + 1) >> 2, c1 = ((idx0 + 1) & 3) * 8;

    v8f acc[4][2];
#pragma unroll
    for (int mi = 0; mi < 4; ++mi)
#pragma unroll
        for (int ni = 0; ni < 2; ++ni) acc[mi][ni] = vzero8();

    // ---- issue async loads of the first K-tile into buffer 0 ----
    async_copy_b128(&A[(size_t)(m0 + r0) * K + c0], &As[0][r0][c0]);
    async_copy_b128(&A[(size_t)(m0 + r1) * K + c1], &As[0][r1][c1]);
    async_copy_b128(&Bm[(size_t)(n0 + r0) * K + c0], &Bs[0][r0][c0]);
    async_copy_b128(&Bm[(size_t)(n0 + r1) * K + c1], &Bs[0][r1][c1]);

    for (int k0 = 0; k0 < K; k0 += 32) {
        const int buf = (k0 >> 5) & 1;
        const bool more = (k0 + 32) < K;
        if (more) {   // prefetch next K-tile into the other buffer (async DMA)
            const int kn = k0 + 32;
            async_copy_b128(&A[(size_t)(m0 + r0) * K + kn + c0], &As[buf ^ 1][r0][c0]);
            async_copy_b128(&A[(size_t)(m0 + r1) * K + kn + c1], &As[buf ^ 1][r1][c1]);
            async_copy_b128(&Bm[(size_t)(n0 + r0) * K + kn + c0], &Bs[buf ^ 1][r0][c0]);
            async_copy_b128(&Bm[(size_t)(n0 + r1) * K + kn + c1], &Bs[buf ^ 1][r1][c1]);
            wait_asynccnt_4();   // async ops complete in order: current tile done
        } else {
            wait_asynccnt_0();
        }
        __syncthreads();

        // --- fragments from LDS ---
        FragBF af[4], bfr[2];
        const int kb = lh * 8;
#pragma unroll
        for (int mi = 0; mi < 4; ++mi) {
            int row = wm * 64 + mi * 16 + l15;
            af[mi].q[0] = *(const uint4*)&As[buf][row][kb];
            af[mi].q[1] = *(const uint4*)&As[buf][row][16 + kb];
        }
        const int kk = lh * 8;
#pragma unroll
        for (int ni = 0; ni < 2; ++ni) {
            int col = wn * 32 + ni * 16 + l15;
            bfr[ni].q[0] = *(const uint4*)&Bs[buf][col][kk];
            bfr[ni].q[1] = *(const uint4*)&Bs[buf][col][16 + kk];
        }
#pragma unroll
        for (int mi = 0; mi < 4; ++mi)
#pragma unroll
            for (int ni = 0; ni < 2; ++ni)
                acc[mi][ni] = wmma_bf16(af[mi].v, bfr[ni].v, acc[mi][ni]);
        __syncthreads();   // guards LDS buffer reuse two iterations out
    }

    // --- epilogue ---
#pragma unroll
    for (int mi = 0; mi < 4; ++mi) {
#pragma unroll
        for (int ni = 0; ni < 2; ++ni) {
#pragma unroll
            for (int r = 0; r < 8; ++r) {
                int row = m0 + wm * 64 + mi * 16 + lh * 8 + r;
                int col = n0 + wn * 32 + ni * 16 + l15;
                float val = acc[mi][ni][r];
                if (MODE == 1) {
                    Cout[(size_t)row * N + col] = val;
                } else {
                    int which = col >> 11;        // 0=q 1=k 2=v
                    int nn    = col & 2047;
                    int h     = nn >> 7;
                    int d     = nn & 127;
                    int b     = row >> 11;        // row / T
                    int t     = row & 2047;       // row % T
                    __bf16* dst = (which == 0) ? Qb : (which == 1) ? Kb : Vb;
                    dst[(((size_t)(b * HH + h)) * TT + t) * DD + d] = (__bf16)val;
                }
            }
        }
    }
}

// ---------------------------------------------------------------------------
// Kernel 3: RoPE (reference style: q1=even dims, q2=odd dims,
// out = [q1*cos - q2*sin, q1*sin + q2*cos]).  One thread per (b,h,t) row,
// reads the whole row before writing -> in-place is race-free.
// ---------------------------------------------------------------------------
__global__ void __launch_bounds__(256) rope_kernel(__bf16* __restrict__ Q,
                                                   __bf16* __restrict__ K,
                                                   const float* __restrict__ freqs) {
    int idx = blockIdx.x * blockDim.x + threadIdx.x;   // over B*H*T rows
    int t   = idx & (TT - 1);
    __bf16* q = Q + (size_t)idx * DD;
    __bf16* k = K + (size_t)idx * DD;

    float s[DD / 2], c[DD / 2];
#pragma unroll 4
    for (int i = 0; i < DD / 2; ++i) {
        float th = (float)t * freqs[i];
        s[i] = __sinf(th);
        c[i] = __cosf(th);
    }
    float buf[DD];
#pragma unroll 4
    for (int i = 0; i < DD; ++i) buf[i] = (float)q[i];
#pragma unroll 4
    for (int i = 0; i < DD / 2; ++i) {
        float e = buf[2 * i], o = buf[2 * i + 1];
        q[i]          = (__bf16)(e * c[i] - o * s[i]);
        q[i + DD / 2] = (__bf16)(e * s[i] + o * c[i]);
    }
#pragma unroll 4
    for (int i = 0; i < DD; ++i) buf[i] = (float)k[i];
#pragma unroll 4
    for (int i = 0; i < DD / 2; ++i) {
        float e = buf[2 * i], o = buf[2 * i + 1];
        k[i]          = (__bf16)(e * c[i] - o * s[i]);
        k[i + DD / 2] = (__bf16)(e * s[i] + o * c[i]);
    }
}

// ---------------------------------------------------------------------------
// Kernel 4: causal flash attention.
// Grid: (T/128, B*H). 256 threads = 8 waves; each wave owns 16 query rows.
// Key tiles of 64 staged in LDS: K via async global->LDS DMA (row-major
// [key][d]); V through VGPRs because it is stored transposed [d][key].
// Online softmax; P written to LDS to re-load in the WMMA A-fragment layout.
// ---------------------------------------------------------------------------
__global__ void __launch_bounds__(256) flash_attn_wmma(
    const __bf16* __restrict__ Q, const __bf16* __restrict__ K,
    const __bf16* __restrict__ V, __bf16* __restrict__ Y) {
    __shared__ __bf16 Ks[64][128];      // 16 KB
    __shared__ __bf16 Vt[128][64];      // 16 KB (transposed: [d][key])
    __shared__ __bf16 Ps[8][16][64];    // 16 KB (per-wave P tiles)

    const int bh   = blockIdx.y;        // b*H + h
    const int b    = bh >> 4;
    const int h    = bh & 15;
    const int qbase = blockIdx.x * 128;
    const int tid  = threadIdx.x;
    const int wave = tid >> 5;
    const int lane = tid & 31;
    const int l15  = lane & 15;
    const int lh   = lane >> 4;
    const int q0   = qbase + wave * 16; // wave's first query row
    const float scale = 0.08838834764831845f;   // 1/sqrt(128)

    const size_t rowbase = (size_t)bh * TT;

    // Q fragments held in registers for the whole kernel (4 x 16x32 over D)
    FragBF qf[4];
    {
        const __bf16* qp = Q + (rowbase + q0 + l15) * DD;
        const int kb = lh * 8;
#pragma unroll
        for (int f = 0; f < 4; ++f) {
            qf[f].q[0] = *(const uint4*)(qp + f * 32 + kb);
            qf[f].q[1] = *(const uint4*)(qp + f * 32 + 16 + kb);
        }
    }

    v8f o[8];
#pragma unroll
    for (int d = 0; d < 8; ++d) o[d] = vzero8();
    float mrow[8], lrow[8];
#pragma unroll
    for (int r = 0; r < 8; ++r) { mrow[r] = -1e30f; lrow[r] = 0.0f; }

    const int nkt = (qbase >> 6) + 2;   // causal: tiles up to the diagonal
    for (int kt = 0; kt < nkt; ++kt) {
        const int kg0 = kt * 64;
        // ---- K tile via async DMA; V tile transposed through VGPRs ----
#pragma unroll
        for (int i = 0; i < 4; ++i) {
            int idx = tid * 4 + i;      // 1024 chunks of 8 bf16
            int kk  = idx >> 4;
            int dc  = (idx & 15) * 8;
            async_copy_b128(&K[(rowbase + kg0 + kk) * DD + dc], &Ks[kk][dc]);
            union { uint4 u; __bf16 e[8]; } vv;
            vv.u = *(const uint4*)&V[(rowbase + kg0 + kk) * DD + dc];
#pragma unroll
            for (int j = 0; j < 8; ++j) Vt[dc + j][kk] = vv.e[j];
        }
        wait_asynccnt_0();
        __syncthreads();

        if (kg0 <= q0 + 15) {           // wave not entirely above the diagonal
            // ---- S = Q K^T : 16 rows x 64 keys (4 accumulators) ----
            v8f s[4];
#pragma unroll
            for (int j = 0; j < 4; ++j) {
                v8f a = vzero8();
                const int key = j * 16 + l15;
#pragma unroll
                for (int f = 0; f < 4; ++f) {
                    FragBF bfr;
                    const int off = f * 32 + lh * 8;
                    bfr.q[0] = *(const uint4*)&Ks[key][off];
                    bfr.q[1] = *(const uint4*)&Ks[key][off + 16];
                    a = wmma_bf16(qf[f].v, bfr.v, a);
                }
                s[j] = a;
            }
            // ---- scale + causal mask + online softmax update ----
            float alpha[8];
#pragma unroll
            for (int r = 0; r < 8; ++r) {
                const int rowg = q0 + lh * 8 + r;
                float tm = -1e30f;
#pragma unroll
                for (int j = 0; j < 4; ++j) {
                    float v = s[j][r] * scale;
                    if (kg0 + j * 16 + l15 > rowg) v = -1e30f;
                    s[j][r] = v;
                    tm = fmaxf(tm, v);
                }
#pragma unroll
                for (int off = 8; off >= 1; off >>= 1)
                    tm = fmaxf(tm, __shfl_xor(tm, off, 16));
                const float mnew = fmaxf(mrow[r], tm);
                alpha[r] = __expf(mrow[r] - mnew);
                float rs = 0.0f;
#pragma unroll
                for (int j = 0; j < 4; ++j) {
                    float v = s[j][r];
                    float p = (v > -1e29f) ? __expf(v - mnew) : 0.0f;
                    Ps[wave][lh * 8 + r][j * 16 + l15] = (__bf16)p;
                    rs += p;
                }
#pragma unroll
                for (int off = 8; off >= 1; off >>= 1)
                    rs += __shfl_xor(rs, off, 16);
                mrow[r] = mnew;
                lrow[r] = lrow[r] * alpha[r] + rs;
            }
            // ---- rescale O ----
#pragma unroll
            for (int d = 0; d < 8; ++d)
#pragma unroll
                for (int r = 0; r < 8; ++r) o[d][r] *= alpha[r];
            // ---- P fragments (A layout) from LDS ----
            FragBF pf[2];
            {
                const int kb = lh * 8;
#pragma unroll
                for (int half = 0; half < 2; ++half) {
                    pf[half].q[0] = *(const uint4*)&Ps[wave][l15][half * 32 + kb];
                    pf[half].q[1] = *(const uint4*)&Ps[wave][l15][half * 32 + 16 + kb];
                }
            }
            // ---- O += P V ----
#pragma unroll
            for (int d = 0; d < 8; ++d) {
                const int col = d * 16 + l15;
#pragma unroll
                for (int half = 0; half < 2; ++half) {
                    FragBF vfr;
                    const int off = half * 32 + lh * 8;
                    vfr.q[0] = *(const uint4*)&Vt[col][off];
                    vfr.q[1] = *(const uint4*)&Vt[col][off + 16];
                    o[d] = wmma_bf16(pf[half].v, vfr.v, o[d]);
                }
            }
        }
        __syncthreads();
    }

    // ---- epilogue: Y[b, t, h*128 + d] = O / l ----
#pragma unroll
    for (int r = 0; r < 8; ++r) {
        const int rowg = q0 + lh * 8 + r;
        const float inv = (lrow[r] > 0.0f) ? (1.0f / lrow[r]) : 0.0f;
        __bf16* yp = Y + ((size_t)b * TT + rowg) * CC + h * DD;
#pragma unroll
        for (int d = 0; d < 8; ++d)
            yp[d * 16 + l15] = (__bf16)(o[d][r] * inv);
    }
}

// ---------------------------------------------------------------------------
// Host launcher
// ---------------------------------------------------------------------------
extern "C" void kernel_launch(void* const* d_in, const int* in_sizes, int n_in,
                              void* d_out, int out_size, void* d_ws, size_t ws_size,
                              hipStream_t stream) {
    (void)in_sizes; (void)n_in; (void)out_size; (void)ws_size;
    const float* x      = (const float*)d_in[0];
    const float* freqs  = (const float*)d_in[1];
    const float* W_attn = (const float*)d_in[2];
    const float* W_proj = (const float*)d_in[3];
    float* out = (float*)d_out;

    const size_t nx  = (size_t)BB * TT * CC;   // 16,777,216
    const size_t nwa = (size_t)3 * CC * CC;    // 12,582,912
    const size_t nwp = (size_t)CC * CC;        //  4,194,304
    const size_t nhd = (size_t)BB * HH * TT * DD;  // == nx

    __bf16* p   = (__bf16*)d_ws;
    __bf16* Xb  = p; p += nx;
    __bf16* Wab = p; p += nwa;
    __bf16* Wpb = p; p += nwp;
    __bf16* Qb  = p; p += nhd;
    __bf16* Kb  = p; p += nhd;
    __bf16* Vb  = p; p += nhd;
    __bf16* Yb  = p; p += nx;   // ~201 MB total workspace

    // 1) convert inputs to bf16
    cvt_f32_bf16<<<dim3((unsigned)((nx  + 255) / 256)), 256, 0, stream>>>(x,      Xb,  nx);
    cvt_f32_bf16<<<dim3((unsigned)((nwa + 255) / 256)), 256, 0, stream>>>(W_attn, Wab, nwa);
    cvt_f32_bf16<<<dim3((unsigned)((nwp + 255) / 256)), 256, 0, stream>>>(W_proj, Wpb, nwp);

    // 2) QKV projection (scatter into [B,H,T,D] bf16)
    gemm_bf16_wmma<0><<<dim3(6144 / 128, 8192 / 128), 256, 0, stream>>>(
        Xb, Wab, nullptr, Qb, Kb, Vb, BB * TT, 3 * CC, CC);

    // 3) RoPE on Q and K
    rope_kernel<<<dim3(BB * HH * TT / 256), 256, 0, stream>>>(Qb, Kb, freqs);

    // 4) causal flash attention -> Y [B,T,C] bf16
    flash_attn_wmma<<<dim3(TT / 128, BB * HH), 256, 0, stream>>>(Qb, Kb, Vb, Yb);

    // 5) output projection -> fp32 d_out
    gemm_bf16_wmma<1><<<dim3(2048 / 128, 8192 / 128), 256, 0, stream>>>(
        Yb, Wpb, out, nullptr, nullptr, nullptr, BB * TT, CC, CC);
}
